// EnhancedXLSTM_78709570667252
// MI455X (gfx1250) — compile-verified, matrix operand order checked by reading, not run on hardware
//
#include <hip/hip_runtime.h>
#include <hip/hip_bf16.h>
#include <cstdint>

typedef __attribute__((ext_vector_type(16))) _Float16 v16h;
typedef __attribute__((ext_vector_type(8)))  _Float16 v8h;
typedef __attribute__((ext_vector_type(8)))  float    v8f;
typedef __attribute__((ext_vector_type(4)))  unsigned int v4u;
typedef __attribute__((ext_vector_type(8)))  int      v8i;
typedef __attribute__((ext_vector_type(4)))  int      v4i;

#define BB 16384
#define DD 512
#define SS 3
#define HH 8
#define HDIM 64
#define LDS_STRIDE 48   /* halves: 96 bytes per row, multiple of 16B for b128 LDS loads */

// ---------------------------------------------------------------------------
// helpers
// ---------------------------------------------------------------------------
__device__ __forceinline__ v8f zero_v8f() {
  v8f z = {0.f, 0.f, 0.f, 0.f, 0.f, 0.f, 0.f, 0.f};
  return z;
}

// Load one 16x32 fp16 fragment (A layout per CDNA5 ISA 7.12.2; B mirrored)
// from an LDS tile whose rows are LDS_STRIDE halves apart.
__device__ __forceinline__ v16h frag_ld(const _Float16* tile, int rowBase, int lane) {
  const int r  = rowBase + (lane & 15);
  const int kb = (lane & 16) ? 8 : 0;   // lanes 16-31 hold K=8..15 / 24..31
  const _Float16* p = tile + r * LDS_STRIDE + kb;
  v8h lo = *reinterpret_cast<const v8h*>(p);        // K = kb .. kb+7
  v8h hi = *reinterpret_cast<const v8h*>(p + 16);   // K = kb+16 .. kb+23
  v16h f;
#pragma unroll
  for (int i = 0; i < 8; ++i) { f[i] = lo[i]; f[i + 8] = hi[i]; }
  return f;
}

__device__ __forceinline__ float sigmf(float x) { return 1.0f / (1.0f + __expf(-x)); }

// Issue one TDM 2-D tile load: tile_rows x 32 halves from row-major source
// (leading dim ld_elems halves) into LDS at ldsByteAddr, with hardware LDS
// padding of 8 DWORDs every 16 DWORDs -> effective LDS row stride 48 halves.
// D# layout per CDNA5 ISA ch.8 (group0 128b, group1 256b, groups2/3 zero).
__device__ __forceinline__ void tdm_load_tile(
    unsigned ldsByteAddr, const void* gsrc, unsigned ld_elems, unsigned tile_rows)
{
  const unsigned long long ga = (unsigned long long)(uintptr_t)gsrc;
  v4u g0;
  g0[0] = 1u;                                          // count=1, is_restore=0, gather=0
  g0[1] = ldsByteAddr;                                 // lds_addr (bytes)
  g0[2] = (unsigned)(ga & 0xffffffffu);                // global_addr[31:0]
  g0[3] = (unsigned)((ga >> 32) & 0x01ffffffu) |       // global_addr[56:32]
          (2u << 30);                                  // type=2 ("image")
  v8i g1;
  g1[0] = (int)((1u << 16)        // data_size = 1 -> 2 bytes
              | (1u << 20)        // pad_enable
              | (3u << 22)        // pad_interval code 3 = 16 DWORDs (= 32 halves)
              | (7u << 25));      // pad_amount  code 7 =  8 DWORDs (= 16 halves)
  g1[1] = 0;                      // atomic_barrier_addr=0, tensor_dim0[15:0]=0
  g1[2] = 0x10;                   // tensor_dim0 = 1M (hi16=0x10); tensor_dim1 lo16=0
  g1[3] = 0x10 | (32 << 16);      // tensor_dim1 = 1M; tile_dim0 = 32 elements
  g1[4] = (int)tile_rows;         // tile_dim1 = rows, tile_dim2 = 0
  g1[5] = (int)ld_elems;          // tensor_dim0_stride[31:0]
  g1[6] = 0;
  g1[7] = 0;
  v4i gz = {0, 0, 0, 0};
  asm volatile("tensor_load_to_lds %0, %1, %2, %3"
               :: "s"(g0), "s"(g1), "s"(gz), "s"(gz)
               : "memory");
}

// ---------------------------------------------------------------------------
// fp32 -> fp16 conversion (grid-stride)
// ---------------------------------------------------------------------------
__global__ void f32_to_f16_kernel(const float* __restrict__ in,
                                  _Float16* __restrict__ out, size_t n) {
  size_t i = (size_t)blockIdx.x * blockDim.x + threadIdx.x;
  size_t stride = (size_t)gridDim.x * blockDim.x;
  for (; i < n; i += stride) out[i] = (_Float16)in[i];
}

// ---------------------------------------------------------------------------
// Generic WMMA GEMM:  C[M x N](f32) = A16[M x K] @ B16[N x K]^T (+bias)(+C)
// block tile 128x64, 256 threads = 8 waves (4 M-waves x 2 N-waves),
// each wave owns a 32x32 tile = 2x2 WMMA fragments, K stepped by 32.
// Double-buffered TDM staging: the Tensor Data Mover streams the next k-slice
// into the alternate LDS buffer while the WMMAs consume the current one.
// ---------------------------------------------------------------------------
__global__ __launch_bounds__(256) void gemm_wmma_kernel(
    const _Float16* __restrict__ A, int lda,
    const _Float16* __restrict__ Bm, int ldb,
    const float* __restrict__ bias,
    float* __restrict__ C, int ldc,
    int K, int accumulate)
{
  __shared__ _Float16 sA[2][128 * LDS_STRIDE];
  __shared__ _Float16 sB[2][64  * LDS_STRIDE];

  const int tid   = threadIdx.x;
  const int lane  = tid & 31;
  const int wave  = __builtin_amdgcn_readfirstlane(tid) >> 5;  // scalar wave id
  const int waveM = wave & 3;
  const int waveN = wave >> 2;
  const int mBase = blockIdx.x * 128;
  const int nBase = blockIdx.y * 64;
  const int m0 = waveM * 32;
  const int n0 = waveN * 32;

  v8f acc[2][2];
#pragma unroll
  for (int i = 0; i < 2; ++i)
#pragma unroll
    for (int j = 0; j < 2; ++j) acc[i][j] = zero_v8f();

  // prologue: DMA first k-slice into buffer 0
  if (wave == 0)
    tdm_load_tile((unsigned)(uintptr_t)sA[0], &A[(size_t)mBase * lda], (unsigned)lda, 128u);
  else if (wave == 1)
    tdm_load_tile((unsigned)(uintptr_t)sB[0], &Bm[(size_t)nBase * ldb], (unsigned)ldb, 64u);

  for (int k0 = 0; k0 < K; k0 += 32) {
    const int cur = (k0 >> 5) & 1;
    if (wave < 2) __builtin_amdgcn_s_wait_tensorcnt(0);
    __syncthreads();                                   // buffers[cur] ready

    if (k0 + 32 < K) {                                 // stream next slice under the math
      if (wave == 0)
        tdm_load_tile((unsigned)(uintptr_t)sA[cur ^ 1],
                      &A[(size_t)mBase * lda + k0 + 32], (unsigned)lda, 128u);
      else if (wave == 1)
        tdm_load_tile((unsigned)(uintptr_t)sB[cur ^ 1],
                      &Bm[(size_t)nBase * ldb + k0 + 32], (unsigned)ldb, 64u);
    }

    v16h a0 = frag_ld(sA[cur], m0, lane);
    v16h a1 = frag_ld(sA[cur], m0 + 16, lane);
    v16h b0 = frag_ld(sB[cur], n0, lane);
    v16h b1 = frag_ld(sB[cur], n0 + 16, lane);

    acc[0][0] = __builtin_amdgcn_wmma_f32_16x16x32_f16(false, a0, false, b0, (short)0, acc[0][0], false, false);
    acc[0][1] = __builtin_amdgcn_wmma_f32_16x16x32_f16(false, a0, false, b1, (short)0, acc[0][1], false, false);
    acc[1][0] = __builtin_amdgcn_wmma_f32_16x16x32_f16(false, a1, false, b0, (short)0, acc[1][0], false, false);
    acc[1][1] = __builtin_amdgcn_wmma_f32_16x16x32_f16(false, a1, false, b1, (short)0, acc[1][1], false, false);
    __syncthreads();                                   // done reading buffers[cur]
  }

  // C/D layout: VGPR v holds M=v (lanes 0-15) / M=v+8 (lanes 16-31), N = lane%16
  const int nCol = lane & 15;
  const int mOff = (lane & 16) ? 8 : 0;
#pragma unroll
  for (int i = 0; i < 2; ++i)
#pragma unroll
    for (int j = 0; j < 2; ++j)
#pragma unroll
      for (int v = 0; v < 8; ++v) {
        const int gr = mBase + m0 + i * 16 + mOff + v;
        const int gc = nBase + n0 + j * 16 + nCol;
        float val = acc[i][j][v];
        if (bias) val += bias[gc];
        const size_t idx = (size_t)gr * ldc + gc;
        if (accumulate) val += C[idx];
        C[idx] = val;
      }
}

// ---------------------------------------------------------------------------
// LayerNorm (over D=512) + exact GELU, writes fp16
// ---------------------------------------------------------------------------
__global__ __launch_bounds__(256) void ln_gelu_kernel(
    const float* __restrict__ xs, const float* __restrict__ g,
    const float* __restrict__ b, _Float16* __restrict__ out)
{
  __shared__ float s1[256], s2[256];
  const int row = blockIdx.x;           // 0 .. S*B-1
  const int s   = row / BB;
  const int tid = threadIdx.x;
  const float* xr = xs + (size_t)row * DD;
  const float x0 = xr[tid];
  const float x1 = xr[tid + 256];
  s1[tid] = x0 + x1;
  s2[tid] = x0 * x0 + x1 * x1;
  __syncthreads();
  for (int off = 128; off > 0; off >>= 1) {
    if (tid < off) { s1[tid] += s1[tid + off]; s2[tid] += s2[tid + off]; }
    __syncthreads();
  }
  const float mean = s1[0] * (1.0f / DD);
  const float var  = s2[0] * (1.0f / DD) - mean * mean;
  const float rinv = rsqrtf(var + 1e-5f);
  const float xv[2] = {x0, x1};
#pragma unroll
  for (int q = 0; q < 2; ++q) {
    const int e = tid + q * 256;
    const float xn = (xv[q] - mean) * rinv * g[s * DD + e] + b[s * DD + e];
    const float ge = 0.5f * xn * (1.0f + erff(xn * 0.70710678118654752f));
    out[(size_t)row * DD + e] = (_Float16)ge;
  }
}

// ---------------------------------------------------------------------------
// Fused gates GEMM + LSTM cell. block = 64 rows x 64 cols of each of the four
// gates (ig/fg/gg/og), 8 waves = 4 gates x 2 row-halves, K = 512 + 512
// (x-part then h-part, flattened so prefetch crosses the phase boundary).
// Double-buffered async global->LDS staging (ASYNCcnt engine); the f32 gates
// tensor (400 MB) never touches HBM.
// ---------------------------------------------------------------------------
__global__ __launch_bounds__(256) void gates_lstm_kernel(
    const _Float16* __restrict__ xs16, const _Float16* __restrict__ hprev16,
    const _Float16* __restrict__ Wih16, const _Float16* __restrict__ Whh16,
    const float* __restrict__ b_ih, const float* __restrict__ b_hh,
    const float* __restrict__ c_prev, const float* __restrict__ decays,
    float* __restrict__ h_new, float* __restrict__ c_new,
    _Float16* __restrict__ hnew16)
{
  __shared__ _Float16 sA[2][64 * LDS_STRIDE];
  __shared__ _Float16 sB[2][4 * 64 * LDS_STRIDE];
  __shared__ float gbuf[4][64][65];

  const int tid  = threadIdx.x;
  const int lane = tid & 31;
  const int wave = __builtin_amdgcn_readfirstlane(tid) >> 5;  // scalar wave id
  const int gate = wave >> 1;          // 0..3 : ig/fg/gg/og
  const int half = wave & 1;           // row half of the 64-row tile
  const int s     = blockIdx.z;
  const int mBase = blockIdx.x * 64;
  const int nBase = blockIdx.y * 64;
  const int m0 = half * 32;

  v8f acc[2][4];
#pragma unroll
  for (int i = 0; i < 2; ++i)
#pragma unroll
    for (int j = 0; j < 4; ++j) acc[i][j] = zero_v8f();

  const int lr = tid >> 2;             // 0..63
  const int lk = (tid & 3) * 8;        // 0,8,16,24

  // per-lane async copy of one 16B chunk of A and of each gate's B tile
  auto issue_async = [&](int buf, int kt) {
    const int k0 = kt & (DD - 1);
    const _Float16* Abase = (kt < DD ? xs16 : hprev16) + (size_t)s * BB * DD;
    const _Float16* Wbase = (kt < DD ? Wih16 : Whh16) + (size_t)s * 4 * DD * DD;
    const _Float16* ga = &Abase[(size_t)(mBase + lr) * DD + k0 + lk];
    asm volatile("global_load_async_to_lds_b128 %0, %1, off"
                 :: "v"((unsigned)(uintptr_t)&sA[buf][lr * LDS_STRIDE + lk]),
                    "v"(ga) : "memory");
#pragma unroll
    for (int g = 0; g < 4; ++g) {
      const _Float16* gb = &Wbase[(size_t)(g * DD + nBase + lr) * DD + k0 + lk];
      asm volatile("global_load_async_to_lds_b128 %0, %1, off"
                   :: "v"((unsigned)(uintptr_t)&sB[buf][g * 64 * LDS_STRIDE + lr * LDS_STRIDE + lk]),
                      "v"(gb) : "memory");
    }
  };

  issue_async(0, 0);                                   // prologue
  for (int kt = 0; kt < 2 * DD; kt += 32) {
    const int cur = (kt >> 5) & 1;
    asm volatile("s_wait_asynccnt 0x0" ::: "memory");
    __syncthreads();                                   // buffers[cur] ready

    if (kt + 32 < 2 * DD) issue_async(cur ^ 1, kt + 32);

    v16h a0 = frag_ld(sA[cur], m0, lane);
    v16h a1 = frag_ld(sA[cur], m0 + 16, lane);
    const _Float16* bt = &sB[cur][gate * 64 * LDS_STRIDE];
#pragma unroll
    for (int j = 0; j < 4; ++j) {
      v16h bf = frag_ld(bt, j * 16, lane);
      acc[0][j] = __builtin_amdgcn_wmma_f32_16x16x32_f16(false, a0, false, bf, (short)0, acc[0][j], false, false);
      acc[1][j] = __builtin_amdgcn_wmma_f32_16x16x32_f16(false, a1, false, bf, (short)0, acc[1][j], false, false);
    }
    __syncthreads();                                   // done reading buffers[cur]
  }

  // add biases, park gate tiles in LDS
  const int nCol = lane & 15;
  const int mOff = (lane & 16) ? 8 : 0;
#pragma unroll
  for (int i = 0; i < 2; ++i)
#pragma unroll
    for (int j = 0; j < 4; ++j)
#pragma unroll
      for (int v = 0; v < 8; ++v) {
        const int ml = m0 + i * 16 + mOff + v;
        const int nl = j * 16 + nCol;
        const int gcol = gate * DD + nBase + nl;
        gbuf[gate][ml][nl] = acc[i][j][v] + b_ih[s * 4 * DD + gcol] + b_hh[s * 4 * DD + gcol];
      }
  __syncthreads();

  // LSTM elementwise: 64x64 elems, 16 per thread
  const float dd  = decays[s];
  const int  row  = tid >> 2;
  const int  cb0  = (tid & 3) * 16;
#pragma unroll
  for (int cc = 0; cc < 16; ++cc) {
    const int col = cb0 + cc;
    const float ig = gbuf[0][row][col];
    const float fg = gbuf[1][row][col];
    const float gg = gbuf[2][row][col];
    const float og = gbuf[3][row][col];
    const size_t gidx = ((size_t)s * BB + mBase + row) * DD + nBase + col;
    const float cp = c_prev[gidx];
    const float cl = sigmf(fg) * cp + sigmf(ig) * tanhf(gg);
    const float hn = sigmf(og) * tanhf(cl);
    const float cn = dd * cp + (1.0f - dd) * cl;
    h_new[gidx]  = hn;
    c_new[gidx]  = cn;
    hnew16[gidx] = (_Float16)hn;
  }
}

// ---------------------------------------------------------------------------
// Attention over S=3 scale-keys per (b,h): one wave32 per (b,h).
// ---------------------------------------------------------------------------
__global__ __launch_bounds__(256) void attn_kernel(
    const float* __restrict__ q, const float* __restrict__ k,
    const float* __restrict__ v, _Float16* __restrict__ fused16)
{
  const int tid  = threadIdx.x;
  const int lane = tid & 31;
  const int wave = tid >> 5;
  const int p = blockIdx.x * 8 + wave;   // (b,h) pair
  const int b = p >> 3;                  // H == 8
  const int h = p & 7;
  const size_t base = (size_t)b * DD + h * HDIM + lane * 2;

  const float2 qv = *reinterpret_cast<const float2*>(&q[base]);
  float sc[SS];
#pragma unroll
  for (int s = 0; s < SS; ++s) {
    const float2 kv = *reinterpret_cast<const float2*>(&k[(size_t)s * BB * DD + base]);
    float part = qv.x * kv.x + qv.y * kv.y;
#pragma unroll
    for (int off = 16; off > 0; off >>= 1) part += __shfl_xor(part, off, 32);
    sc[s] = part * 0.125f;               // 1/sqrt(64)
  }
  const float mx = fmaxf(sc[0], fmaxf(sc[1], sc[2]));
  const float e0 = __expf(sc[0] - mx), e1 = __expf(sc[1] - mx), e2 = __expf(sc[2] - mx);
  const float inv = 1.0f / (e0 + e1 + e2);
  const float w[SS] = {e0 * inv, e1 * inv, e2 * inv};
  float o0 = 0.f, o1 = 0.f;
#pragma unroll
  for (int s = 0; s < SS; ++s) {
    const float2 vv = *reinterpret_cast<const float2*>(&v[(size_t)s * BB * DD + base]);
    o0 += w[s] * vv.x;
    o1 += w[s] * vv.y;
  }
  fused16[base]     = (_Float16)o0;
  fused16[base + 1] = (_Float16)o1;
}

__global__ void bias_combine_kernel(const float* __restrict__ a,
                                    const float* __restrict__ b,
                                    float* __restrict__ o) {
  const int i = blockIdx.x * blockDim.x + threadIdx.x;
  if (i < DD) o[i] = a[i] + b[i];
}

// ---------------------------------------------------------------------------
// launch
// ---------------------------------------------------------------------------
extern "C" void kernel_launch(void* const* d_in, const int* in_sizes, int n_in,
                              void* d_out, int out_size, void* d_ws, size_t ws_size,
                              hipStream_t stream)
{
  (void)in_sizes; (void)n_in; (void)out_size; (void)ws_size;

  const float* x      = (const float*)d_in[0];
  const float* h_prev = (const float*)d_in[1];
  const float* c_prev = (const float*)d_in[2];
  const float* ssm    = (const float*)d_in[3];
  const float* Wt     = (const float*)d_in[4];
  const float* bt     = (const float*)d_in[5];
  const float* ln_g   = (const float*)d_in[6];
  const float* ln_b   = (const float*)d_in[7];
  const float* W_ih   = (const float*)d_in[8];
  const float* W_hh   = (const float*)d_in[9];
  const float* b_ih   = (const float*)d_in[10];
  const float* b_hh   = (const float*)d_in[11];
  const float* decays = (const float*)d_in[12];
  const float* inpw   = (const float*)d_in[13];
  const float* inpb   = (const float*)d_in[14];
  const float* outw   = (const float*)d_in[15];
  const float* outb   = (const float*)d_in[16];
  const float* mixw   = (const float*)d_in[17];
  const float* mixb   = (const float*)d_in[18];

  float* out   = (float*)d_out;
  float* h_new = out + (size_t)BB * DD;
  float* c_new = h_new + (size_t)SS * BB * DD;

  char* wsb = (char*)d_ws;
  size_t off = 0;
  auto alloc = [&](size_t bytes) -> char* {
    char* p = wsb + off;
    off += (bytes + 255) & ~(size_t)255;
    return p;
  };
  _Float16* x16     = (_Float16*)alloc((size_t)BB * DD * 2);
  _Float16* Wt16    = (_Float16*)alloc((size_t)SS * DD * DD * 2);
  _Float16* hprev16 = (_Float16*)alloc((size_t)SS * BB * DD * 2);
  _Float16* Wih16   = (_Float16*)alloc((size_t)SS * 4 * DD * DD * 2);
  _Float16* Whh16   = (_Float16*)alloc((size_t)SS * 4 * DD * DD * 2);
  _Float16* ssm16   = (_Float16*)alloc((size_t)BB * DD * 2);
  _Float16* inp16   = (_Float16*)alloc((size_t)3 * DD * DD * 2);
  _Float16* outw16  = (_Float16*)alloc((size_t)DD * DD * 2);
  _Float16* mix16   = (_Float16*)alloc((size_t)DD * 3 * DD * 2);
  float*    xsf     = (float*)   alloc((size_t)SS * BB * DD * 4);
  _Float16* xs16    = (_Float16*)alloc((size_t)SS * BB * DD * 2);
  _Float16* hnew16  = (_Float16*)alloc((size_t)SS * BB * DD * 2);
  float*    q32     = (float*)   alloc((size_t)BB * DD * 4);
  float*    k32     = (float*)   alloc((size_t)SS * BB * DD * 4);
  float*    v32     = (float*)   alloc((size_t)SS * BB * DD * 4);
  _Float16* fused16 = (_Float16*)alloc((size_t)BB * DD * 2);
  float*    cbias   = (float*)   alloc((size_t)DD * 4);

  auto cvt = [&](const float* src, _Float16* dst, size_t n) {
    f32_to_f16_kernel<<<2048, 256, 0, stream>>>(src, dst, n);
  };
  cvt(x,      x16,     (size_t)BB * DD);
  cvt(Wt,     Wt16,    (size_t)SS * DD * DD);
  cvt(h_prev, hprev16, (size_t)SS * BB * DD);
  cvt(W_ih,   Wih16,   (size_t)SS * 4 * DD * DD);
  cvt(W_hh,   Whh16,   (size_t)SS * 4 * DD * DD);
  cvt(ssm,    ssm16,   (size_t)BB * DD);
  cvt(inpw,   inp16,   (size_t)3 * DD * DD);
  cvt(outw,   outw16,  (size_t)DD * DD);
  cvt(mixw,   mix16,   (size_t)DD * 3 * DD);

  const dim3 gGemm(BB / 128, DD / 64);

  // xs = x @ Wt[s]^T + bt[s]
  for (int s = 0; s < SS; ++s)
    gemm_wmma_kernel<<<gGemm, 256, 0, stream>>>(
        x16, DD, Wt16 + (size_t)s * DD * DD, DD,
        bt + s * DD, xsf + (size_t)s * BB * DD, DD, DD, 0);

  // LayerNorm + GELU -> xs16
  ln_gelu_kernel<<<SS * BB, 256, 0, stream>>>(xsf, ln_g, ln_b, xs16);

  // gates GEMM fused with LSTM cell
  gates_lstm_kernel<<<dim3(BB / 64, DD / 64, SS), 256, 0, stream>>>(
      xs16, hprev16, Wih16, Whh16, b_ih, b_hh, c_prev, decays,
      h_new, c_new, hnew16);

  // q = ssm_state @ Wq^T + bq
  gemm_wmma_kernel<<<gGemm, 256, 0, stream>>>(
      ssm16, DD, inp16, DD, inpb, q32, DD, DD, 0);

  // k[s], v[s] = h_new[s] @ Wk^T / Wv^T
  for (int s = 0; s < SS; ++s) {
    gemm_wmma_kernel<<<gGemm, 256, 0, stream>>>(
        hnew16 + (size_t)s * BB * DD, DD, inp16 + (size_t)DD * DD, DD,
        inpb + DD, k32 + (size_t)s * BB * DD, DD, DD, 0);
    gemm_wmma_kernel<<<gGemm, 256, 0, stream>>>(
        hnew16 + (size_t)s * BB * DD, DD, inp16 + (size_t)2 * DD * DD, DD,
        inpb + 2 * DD, v32 + (size_t)s * BB * DD, DD, DD, 0);
  }

  // softmax attention over S=3
  attn_kernel<<<BB * HH / 8, 256, 0, stream>>>(q32, k32, v32, fused16);

  // out = fused @ out_proj^T + (out_proj_b + mix_b) + sum_s h_new[s] @ mix_w[:, sD:(s+1)D]^T
  bias_combine_kernel<<<2, 256, 0, stream>>>(outb, mixb, cbias);
  gemm_wmma_kernel<<<gGemm, 256, 0, stream>>>(
      fused16, DD, outw16, DD, cbias, out, DD, DD, 0);
  for (int s = 0; s < SS; ++s)
    gemm_wmma_kernel<<<gGemm, 256, 0, stream>>>(
        hnew16 + (size_t)s * BB * DD, DD, mix16 + s * DD, 3 * DD,
        nullptr, out, DD, DD, 1);
}
